// MixtureOfExperts_38774964748492
// MI455X (gfx1250) — matrix operand order repaired and checked by reading, not compiled, not run on hardware
//
#include <hip/hip_runtime.h>
#include <hip/hip_bf16.h>

typedef __attribute__((ext_vector_type(16))) __bf16 v16bf;
typedef __attribute__((ext_vector_type(8)))  __bf16 v8bf;
typedef __attribute__((ext_vector_type(8)))  float  v8f;
typedef int v4i_vs __attribute__((vector_size(16)));   // matches async-LDS builtin pointee

#define D_DIM  1024
#define H_DIM  1024
#define N_EXP  8
#define N_TOK  1024
#define N_DISP 2048

// LDS B-panel: 64 columns x 1024 K, row stride padded to 1032 bf16 (2064 B)
#define BPAD   1032
#define LDS_BYTES (64 * BPAD * 2)

#if defined(__gfx1250__) && __has_builtin(__builtin_amdgcn_global_load_async_to_lds_b128)
#define USE_ASYNC_LDS 1
#else
#define USE_ASYNC_LDS 0
#endif
#define AS1 __attribute__((address_space(1)))
#define AS3 __attribute__((address_space(3)))

#define WMMA_BF16(a, b, c) __builtin_amdgcn_wmma_f32_16x16x32_bf16(false, a, false, b, (short)0, c, false, false)

// ---------------- prep: fp32 -> bf16 copy of x ----------------
__global__ void k_convert_x(const float* __restrict__ x, __bf16* __restrict__ xb) {
    int i = (blockIdx.x * blockDim.x + threadIdx.x) * 4;
    float4 v = *(const float4*)(x + i);
    xb[i + 0] = (__bf16)v.x;
    xb[i + 1] = (__bf16)v.y;
    xb[i + 2] = (__bf16)v.z;
    xb[i + 3] = (__bf16)v.w;
}

// ---------------- prep: transpose+convert weights to [E][N][K] bf16 ----------------
__global__ void k_transpose_w(const float* __restrict__ w_in, const float* __restrict__ w_out,
                              __bf16* __restrict__ w1t, __bf16* __restrict__ w2t) {
    __shared__ float tile[32][33];
    const int z = blockIdx.z;
    const float* src;
    __bf16* dst;
    if (z < N_EXP) {
        src = w_in + (size_t)z * D_DIM * H_DIM;
        dst = w1t  + (size_t)z * D_DIM * H_DIM;
    } else {
        src = w_out + (size_t)(z - N_EXP) * H_DIM * D_DIM;
        dst = w2t   + (size_t)(z - N_EXP) * H_DIM * D_DIM;
    }
    const int r0 = blockIdx.y * 32;
    const int c0 = blockIdx.x * 32;
    for (int rr = threadIdx.y; rr < 32; rr += 8)
        tile[rr][threadIdx.x] = src[(size_t)(r0 + rr) * 1024 + c0 + threadIdx.x];
    __syncthreads();
    for (int rr = threadIdx.y; rr < 32; rr += 8)
        dst[(size_t)(c0 + rr) * 1024 + r0 + threadIdx.x] = (__bf16)tile[threadIdx.x][rr];
}

__global__ void k_zero_cnt(int* cnt) {
    if (threadIdx.x < N_EXP) cnt[threadIdx.x] = 0;
}

// ---------------- router: logits, softmax, top-2, expert compaction ----------------
__global__ void k_router(const float* __restrict__ x, const float* __restrict__ rw,
                         float* __restrict__ top_p, int* __restrict__ cnt,
                         int* __restrict__ rows) {
    __shared__ float srw[D_DIM * N_EXP];           // 32 KB
    for (int i = threadIdx.x; i < D_DIM * N_EXP; i += blockDim.x) srw[i] = rw[i];
    __syncthreads();

    const int t = blockIdx.x * blockDim.x + threadIdx.x;
    const float* xr = x + (size_t)t * D_DIM;
    float l[N_EXP];
#pragma unroll
    for (int j = 0; j < N_EXP; ++j) l[j] = 0.f;
    for (int d = 0; d < D_DIM; ++d) {
        float xv = xr[d];
#pragma unroll
        for (int j = 0; j < N_EXP; ++j) l[j] += xv * srw[d * N_EXP + j];
    }
    float mx = l[0];
#pragma unroll
    for (int j = 1; j < N_EXP; ++j) mx = fmaxf(mx, l[j]);
    float p[N_EXP], sum = 0.f;
#pragma unroll
    for (int j = 0; j < N_EXP; ++j) { p[j] = __expf(l[j] - mx); sum += p[j]; }
    float inv = 1.f / sum;
    int e0 = 0; float b0 = l[0];
#pragma unroll
    for (int j = 1; j < N_EXP; ++j) if (l[j] > b0) { b0 = l[j]; e0 = j; }
    int e1 = -1; float b1 = -3.4e38f;
#pragma unroll
    for (int j = 0; j < N_EXP; ++j) if (j != e0 && l[j] > b1) { b1 = l[j]; e1 = j; }

    const int   ek[2] = { e0, e1 };
    const float pk[2] = { p[e0] * inv, p[e1] * inv };
#pragma unroll
    for (int k = 0; k < 2; ++k) {
        int id  = t * 2 + k;
        int pos = atomicAdd(&cnt[ek[k]], 1);
        rows[ek[k] * N_DISP + pos] = id;
        top_p[id] = pk[k];
    }
}

__device__ __forceinline__ float gelu_tanh(float v) {
    float c = 0.7978845608028654f * (v + 0.044715f * v * v * v);
    return 0.5f * v * (1.f + tanhf(c));
}

// load 16 bf16 as two 16B chunks (p, p+off2) -> one WMMA operand fragment
__device__ __forceinline__ v16bf ld_pair(const __bf16* p, int off2) {
    v8bf t0 = *(const v8bf*)p;
    v8bf t1 = *(const v8bf*)(p + off2);
    return __builtin_shufflevector(t0, t1, 0, 1, 2, 3, 4, 5, 6, 7,
                                           8, 9, 10, 11, 12, 13, 14, 15);
}

// ---------------- expert GEMMs via bf16 WMMA, B panel staged in LDS ----------------
// grid: (Mtiles=16, Ntiles=16, E=8), block (32,8). Wave strip: 16 rows x 64 cols.
template <bool FFN1>
__global__ __launch_bounds__(256) void k_ffn(
        const __bf16* __restrict__ A, const __bf16* __restrict__ Wt,
        const float* __restrict__ bias, const float* __restrict__ top_p,
        const int* __restrict__ cnt, const int* __restrict__ rows,
        __bf16* __restrict__ hb, float* __restrict__ yd) {
    extern __shared__ __bf16 lds_b[];           // [64][BPAD]
    const int e  = blockIdx.z;
    const int ne = cnt[e];
    const int m0 = blockIdx.x * 128;
    if (m0 >= ne) return;                       // uniform: whole block exits together
    const int n0   = blockIdx.y * 64;
    const int lane = threadIdx.x;
    const int base = m0 + threadIdx.y * 16;
    const int* rlist = rows + e * N_DISP;

    // ---- stage contiguous 64x1024 bf16 weight panel (128 KB) into LDS ----
    const __bf16* wpanel = Wt + (size_t)e * 1024 * 1024 + (size_t)n0 * 1024;
    const int tid = threadIdx.y * 32 + threadIdx.x;
    for (int c = tid; c < 8192; c += 256) {     // 8192 chunks of 8 bf16 (16 B)
        int nl = c >> 7, kc = c & 127;
#if USE_ASYNC_LDS
        __builtin_amdgcn_global_load_async_to_lds_b128(
            (AS1 v4i_vs*)(wpanel + (size_t)nl * 1024 + kc * 8),
            (AS3 v4i_vs*)(lds_b + nl * BPAD + kc * 8), 0, 0);
#else
        *(v8bf*)(lds_b + nl * BPAD + kc * 8) =
            *(const v8bf*)(wpanel + (size_t)nl * 1024 + kc * 8);
#endif
    }
#if USE_ASYNC_LDS
#if __has_builtin(__builtin_amdgcn_s_wait_asynccnt)
    __builtin_amdgcn_s_wait_asynccnt(0);
#else
    asm volatile("s_wait_asynccnt 0x0" ::: "memory");
#endif
#endif
    __syncthreads();

    // A fragment row for this lane (clamped gather; masked at store)
    int ra = base + (lane & 15);
    int rid_a = rlist[ra < ne ? ra : (ne - 1)];
    const __bf16* arow = A + (size_t)(FFN1 ? (rid_a >> 1) : rid_a) * 1024;
    const int ks   = (lane >= 16) ? 16 : 0;     // A layout: lanes 16-31 hold K 16..31
    const int koff = (lane >= 16) ? 16 : 0;     // B layout: lanes 16-31 hold K 16..31
    const int nlo  = lane & 15;
    const __bf16* b0p = lds_b + (0 * 16 + nlo) * BPAD + koff;
    const __bf16* b1p = lds_b + (1 * 16 + nlo) * BPAD + koff;
    const __bf16* b2p = lds_b + (2 * 16 + nlo) * BPAD + koff;
    const __bf16* b3p = lds_b + (3 * 16 + nlo) * BPAD + koff;

    v8f acc[4];
#pragma unroll
    for (int j = 0; j < 4; ++j) acc[j] = 0.f;

    // two-stage manual double buffer: named sets, no rotation, no unroll
    v16bf aX  = ld_pair(arow + ks, 8);
    v16bf bX0 = ld_pair(b0p, 8), bX1 = ld_pair(b1p, 8);
    v16bf bX2 = ld_pair(b2p, 8), bX3 = ld_pair(b3p, 8);
    v16bf aY, bY0, bY1, bY2, bY3;
#pragma clang loop unroll(disable)
    for (int k0 = 0; k0 < 960; k0 += 64) {      // 15 iters: steps k0 and k0+32
        aY  = ld_pair(arow + k0 + 32 + ks, 8);
        bY0 = ld_pair(b0p + k0 + 32, 8);
        bY1 = ld_pair(b1p + k0 + 32, 8);
        bY2 = ld_pair(b2p + k0 + 32, 8);
        bY3 = ld_pair(b3p + k0 + 32, 8);
        acc[0] = WMMA_BF16(aX, bX0, acc[0]);
        acc[1] = WMMA_BF16(aX, bX1, acc[1]);
        acc[2] = WMMA_BF16(aX, bX2, acc[2]);
        acc[3] = WMMA_BF16(aX, bX3, acc[3]);
        aX  = ld_pair(arow + k0 + 64 + ks, 8);
        bX0 = ld_pair(b0p + k0 + 64, 8);
        bX1 = ld_pair(b1p + k0 + 64, 8);
        bX2 = ld_pair(b2p + k0 + 64, 8);
        bX3 = ld_pair(b3p + k0 + 64, 8);
        acc[0] = WMMA_BF16(aY, bY0, acc[0]);
        acc[1] = WMMA_BF16(aY, bY1, acc[1]);
        acc[2] = WMMA_BF16(aY, bY2, acc[2]);
        acc[3] = WMMA_BF16(aY, bY3, acc[3]);
    }
    // tail: steps k=960 (in X) and k=992 (load into Y)
    aY  = ld_pair(arow + 992 + ks, 8);
    bY0 = ld_pair(b0p + 992, 8);
    bY1 = ld_pair(b1p + 992, 8);
    bY2 = ld_pair(b2p + 992, 8);
    bY3 = ld_pair(b3p + 992, 8);
    acc[0] = WMMA_BF16(aX, bX0, acc[0]);
    acc[1] = WMMA_BF16(aX, bX1, acc[1]);
    acc[2] = WMMA_BF16(aX, bX2, acc[2]);
    acc[3] = WMMA_BF16(aX, bX3, acc[3]);
    acc[0] = WMMA_BF16(aY, bY0, acc[0]);
    acc[1] = WMMA_BF16(aY, bY1, acc[1]);
    acc[2] = WMMA_BF16(aY, bY2, acc[2]);
    acc[3] = WMMA_BF16(aY, bY3, acc[3]);

    // epilogue: C layout — VGPR v, lanes 0-15: M=v, N=lane; lanes 16-31: M=8+v, N=lane-16
    const int mhi = (lane >= 16) ? 8 : 0;
    int   rid[8];
    float pw[8];
#pragma unroll
    for (int v = 0; v < 8; ++v) {
        int r = base + mhi + v;
        rid[v] = (r < ne) ? rlist[r] : -1;
        if (!FFN1) pw[v] = (rid[v] >= 0) ? top_p[rid[v]] : 0.f;
    }
#pragma unroll
    for (int j = 0; j < 4; ++j) {
        int n = n0 + j * 16 + nlo;
        float bv = bias[e * 1024 + n];
#pragma unroll
        for (int v = 0; v < 8; ++v) {
            if (rid[v] >= 0) {
                float val = acc[j][v] + bv;
                if (FFN1)
                    hb[(size_t)rid[v] * 1024 + n] = (__bf16)gelu_tanh(val);
                else
                    yd[(size_t)rid[v] * 1024 + n] = val * pw[v];
            }
        }
    }
}

// ---------------- deterministic combine: out[t] = yd[2t] + yd[2t+1] ----------------
__global__ void k_combine(const float* __restrict__ yd, float* __restrict__ out) {
    int idx = blockIdx.x * blockDim.x + threadIdx.x;
    int t = idx >> 10, d = idx & 1023;
    size_t r0 = (size_t)(2 * t) * 1024 + d;
    out[idx] = yd[r0] + yd[r0 + 1024];
}

extern "C" void kernel_launch(void* const* d_in, const int* in_sizes, int n_in,
                              void* d_out, int out_size, void* d_ws, size_t ws_size,
                              hipStream_t stream) {
    const float* x     = (const float*)d_in[0];
    const float* rw    = (const float*)d_in[1];
    const float* w_in  = (const float*)d_in[2];
    const float* b_in  = (const float*)d_in[3];
    const float* w_out = (const float*)d_in[4];
    const float* b_out = (const float*)d_in[5];
    float* out = (float*)d_out;

    char* ws = (char*)d_ws;
    size_t off = 0;
    auto alloc = [&](size_t bytes) -> void* {
        void* p = ws + off;
        off += (bytes + 255) & ~(size_t)255;
        return p;
    };
    __bf16* xb   = (__bf16*)alloc((size_t)N_TOK * D_DIM * 2);
    __bf16* w1t  = (__bf16*)alloc((size_t)N_EXP * H_DIM * D_DIM * 2);
    __bf16* w2t  = (__bf16*)alloc((size_t)N_EXP * D_DIM * H_DIM * 2);
    __bf16* hb   = (__bf16*)alloc((size_t)N_DISP * H_DIM * 2);
    float*  yd   = (float*) alloc((size_t)N_DISP * D_DIM * 4);
    float*  topp = (float*) alloc((size_t)N_DISP * 4);
    int*    cnt  = (int*)   alloc(N_EXP * 4);
    int*    rows = (int*)   alloc((size_t)N_EXP * N_DISP * 4);
    (void)ws_size; (void)in_sizes; (void)n_in; (void)out_size;

    k_zero_cnt<<<1, 32, 0, stream>>>(cnt);
    k_convert_x<<<(N_TOK * D_DIM) / (4 * 256), 256, 0, stream>>>(x, xb);
    k_transpose_w<<<dim3(32, 32, 16), dim3(32, 8), 0, stream>>>(w_in, w_out, w1t, w2t);
    k_router<<<N_TOK / 128, 128, 0, stream>>>(x, rw, topp, cnt, rows);
    k_ffn<true ><<<dim3(16, 16, N_EXP), dim3(32, 8), LDS_BYTES, stream>>>(xb, w1t, b_in,  topp, cnt, rows, hb, yd);
    k_ffn<false><<<dim3(16, 16, N_EXP), dim3(32, 8), LDS_BYTES, stream>>>(hb, w2t, b_out, topp, cnt, rows, hb, yd);
    k_combine<<<(N_TOK * D_DIM) / 256, 256, 0, stream>>>(yd, out);
}
